// RNN_10453950398514
// MI455X (gfx1250) — matrix operand order repaired and checked by reading, not compiled, not run on hardware
//
#include <hip/hip_runtime.h>
#include <cstdint>
#include <cstddef>

namespace {
constexpr int kIn   = 57;           // INPUT_SIZE
constexpr int kHid  = 64;           // HIDDEN_SIZE
constexpr int kOut  = 7;            // OUTPUT_SIZE
constexpr int kT    = 128;          // sequence length
constexpr int kK    = kIn + kHid;   // 121 combined features
constexpr int kS    = 68;           // fp32 LDS row stride (bank-conflict free)
constexpr int kS16  = 72;           // bf16 LDS row stride (144B: 16B-aligned, conflict-free)
}

typedef __attribute__((ext_vector_type(16))) __bf16          v16bf;
typedef __attribute__((ext_vector_type(16))) unsigned short  v16u;
typedef __attribute__((ext_vector_type(8)))  unsigned short  v8u;
typedef __attribute__((ext_vector_type(8)))  float           v8f;

__device__ __forceinline__ unsigned short f32_to_bf16_rne(float f) {
  uint32_t u = __builtin_bit_cast(uint32_t, f);
  u += 0x7FFFu + ((u >> 16) & 1u);          // round-to-nearest-even
  return (unsigned short)(u >> 16);
}

// Load one 32-wide K-chunk of the 16-bit A fragment from a bf16 LDS row:
// lane half hi: elements j<8 -> row[8*hi + j], j>=8 -> row[16 + 8*hi + (j-8)].
// Two aligned 16B loads (ds_load_b128), zero conversion VALU.
__device__ __forceinline__ v16bf load_a_chunk16(const unsigned short* row, int hi) {
  const v8u* p0 = (const v8u*)(row + 8 * hi);
  const v8u* p1 = (const v8u*)(row + 16 + 8 * hi);
  union { v8u h[2]; v16u v; } u;
  u.h[0] = p0[0];
  u.h[1] = p1[0];
  return __builtin_bit_cast(v16bf, u.v);
}

// Branchless bit-select: (a & sel) | (b & ~sel) -> single v_bfi_b32.
__device__ __forceinline__ float bsel(uint32_t sel, float a, float b) {
  const uint32_t r = (__builtin_bit_cast(uint32_t, a) & sel) |
                     (__builtin_bit_cast(uint32_t, b) & ~sel);
  return __builtin_bit_cast(float, r);
}

// Block = 64 threads = 2 waves per 16-row batch tile. Wave w owns N-tiles
// {2w, 2w+1}: W_i2h fragments cost 64 persistent VGPRs per wave. Hidden state
// is fp32 in LDS with a bf16 shadow copy in the WMMA A layout; each step runs
// 8 v_wmma_f32_16x16x32_bf16 per wave with f32 accumulation, 2 barriers/step.
__global__ __launch_bounds__(64) void rnn_wmma_kernel(
    const float* __restrict__ x,   // [B, T, 57]
    const float* __restrict__ Wh,  // [64, 121]
    const float* __restrict__ bh,  // [64]
    const float* __restrict__ Wo,  // [7, 121]
    const float* __restrict__ bo,  // [7]
    float* __restrict__ out) {     // [B, 7]
  __shared__ float sh[16][kS];                      // fp32 hidden state
  __shared__ float slasth[16][kS];                  // h before last valid update
  __shared__ __align__(16) unsigned short sh16[16][kS16];  // bf16 shadow of h
  __shared__ __align__(16) unsigned short sx16[16][kS16];  // bf16 staged x (k>=57 stays 0)
  __shared__ float sv[16];                          // raw fp32 x[m][0] of current step
  __shared__ int   lastt[16];                       // last valid timestep per row (-1 = none)
  __shared__ float sWo[kOut][kK + 3];

  const int tid  = threadIdx.x;     // 0..63
  const int w    = tid >> 5;        // wave id: owns N-tiles 2w, 2w+1
  const int lane = tid & 31;
  const int lo   = lane & 15;
  const int hi   = lane >> 4;       // lane half selects K sub-range per ISA layout
  const int b0   = blockIdx.x * 16;

  for (int i = tid; i < 16 * kS; i += 64) { (&sh[0][0])[i] = 0.f; (&slasth[0][0])[i] = 0.f; }
  for (int i = tid; i < 16 * kS16; i += 64) { (&sh16[0][0])[i] = 0; (&sx16[0][0])[i] = 0; }
  if (tid < 16) lastt[tid] = -1;
  for (int i = tid; i < kOut * kK; i += 64) sWo[i / kK][i % kK] = Wo[i];

  // ---- Loop-invariant B fragments: B[k][n] = W_i2h[n][combined(k)] ----
  // Padded K space: k<57 -> x weights; 57..63 -> zero; 64..127 -> h weights (k-7).
  v16bf Bf[2][4];                   // [local ntile][kchunk] -> 64 persistent VGPRs
  #pragma unroll
  for (int nt = 0; nt < 2; ++nt) {
    const int n = (2 * w + nt) * 16 + lo;
    #pragma unroll
    for (int c = 0; c < 4; ++c) {
      v16u tmp;
      #pragma unroll
      for (int j = 0; j < 16; ++j) {
        const int k = c * 32 + (j < 8 ? 8 * hi + j : 16 + 8 * hi + (j - 8));
        float wv = 0.f;
        if (k < kIn)      wv = Wh[n * kK + k];
        else if (k >= 64) wv = Wh[n * kK + (k - 7)];
        tmp[j] = f32_to_bf16_rne(wv);
      }
      Bf[nt][c] = __builtin_bit_cast(v16bf, tmp);
    }
  }
  float bias[2];
  #pragma unroll
  for (int nt = 0; nt < 2; ++nt) bias[nt] = bh[(2 * w + nt) * 16 + lo];

  __syncthreads();   // zero-init visible before staging writes

  // Stage step t: bf16-convert the 16x57 x tile into sx16; keep raw x[m][0] in sv;
  // track last valid t; prefetch step t+1.
  auto stage = [&](int t) {
    const int m  = tid & 15;
    const int k0 = tid >> 4;                        // 4 lanes per row
    const float* xr = x + ((size_t)(b0 + m) * kT + t) * kIn;
    #pragma unroll
    for (int kk = 0; kk < 60; kk += 4) {
      const int k = kk + k0;
      if (k < kIn) sx16[m][k] = f32_to_bf16_rne(xr[k]);
    }
    if (tid < 16) {
      const float v0 = xr[0];
      sv[tid] = v0;
      const int tt = lastt[tid];
      lastt[tid] = (v0 != -1.0f) ? t : tt;          // branchless cndmask
    }
    if (t + 1 < kT && tid < 32) {                   // global_prefetch_b8 next step
      __builtin_prefetch(x + ((size_t)(b0 + (tid & 15)) * kT + t + 1) * kIn + (tid >> 4) * 33, 0, 0);
    }
  };

  stage(0);                                         // prologue
  __syncthreads();

  for (int t = 0; t < kT; ++t) {
    // ---- Phase B: capture valid masks, load pre-packed bf16 A fragments, WMMA.
    uint32_t selm[8];                               // all-ones if row valid
    #pragma unroll
    for (int v = 0; v < 8; ++v)
      selm[v] = (sv[v + 8 * hi] != -1.0f) ? 0xFFFFFFFFu : 0u;

    const v16bf a0 = load_a_chunk16(&sx16[lo][0],  hi);  // K   0..31  (x)
    const v16bf a1 = load_a_chunk16(&sx16[lo][32], hi);  // K  32..63  (x + zero pad)
    const v16bf a2 = load_a_chunk16(&sh16[lo][0],  hi);  // K  64..95  (h)
    const v16bf a3 = load_a_chunk16(&sh16[lo][32], hi);  // K  96..127 (h)

    v8f d[2];
    #pragma unroll
    for (int nt = 0; nt < 2; ++nt) {
      v8f c;
      #pragma unroll
      for (int v = 0; v < 8; ++v) c[v] = bias[nt];
      c = __builtin_amdgcn_wmma_f32_16x16x32_bf16(false, a0, false, Bf[nt][0], (short)0, c, false, false);
      c = __builtin_amdgcn_wmma_f32_16x16x32_bf16(false, a1, false, Bf[nt][1], (short)0, c, false, false);
      c = __builtin_amdgcn_wmma_f32_16x16x32_bf16(false, a2, false, Bf[nt][2], (short)0, c, false, false);
      c = __builtin_amdgcn_wmma_f32_16x16x32_bf16(false, a3, false, Bf[nt][3], (short)0, c, false, false);
      d[nt] = c;
    }
    __syncthreads();   // all reads of sx16/sh16 for step t done

    // ---- Phase C: branchless writeback (v_bfi_b32 selects force eager loads)
    //      plus pipelined staging of t+1. D layout: lane holds n = lo, m = v + 8*hi.
    #pragma unroll
    for (int nt = 0; nt < 2; ++nt) {
      const int n = (2 * w + nt) * 16 + lo;
      #pragma unroll
      for (int v = 0; v < 8; ++v) {
        const int m = v + 8 * hi;
        const float oldh = sh[m][n];
        const float oldl = slasth[m][n];
        const float newh = bsel(selm[v], d[nt][v], oldh);  // v_bfi_b32
        slasth[m][n] = bsel(selm[v], oldh, oldl);          // v_bfi_b32
        sh[m][n]     = newh;
        sh16[m][n]   = f32_to_bf16_rne(newh);              // bf16 shadow for next step
      }
    }
    if (t + 1 < kT) stage(t + 1);
    __syncthreads();   // writeback + new sx16/sv visible before next step
  }

  // ---- Deferred output head, full fp32: log_softmax([x_{t*}, h_{t*-1}] @ Wo^T + bo).
  if (tid < 16) {
    const int m  = tid;
    const int tt = lastt[m];
    if (tt >= 0) {
      const float* xr = x + ((size_t)(b0 + m) * kT + tt) * kIn;
      float z[kOut];
      #pragma unroll
      for (int o = 0; o < kOut; ++o) {
        float acc = bo[o];
        for (int k = 0; k < kIn; ++k)  acc = __builtin_fmaf(xr[k],        sWo[o][k],       acc);
        for (int j = 0; j < kHid; ++j) acc = __builtin_fmaf(slasth[m][j], sWo[o][kIn + j], acc);
        z[o] = acc;
      }
      float mx = z[0];
      #pragma unroll
      for (int o = 1; o < kOut; ++o) mx = fmaxf(mx, z[o]);
      float s = 0.f;
      #pragma unroll
      for (int o = 0; o < kOut; ++o) s += __expf(z[o] - mx);
      const float lse = __logf(s);
      #pragma unroll
      for (int o = 0; o < kOut; ++o) out[(size_t)(b0 + m) * kOut + o] = z[o] - mx - lse;
    } else {
      #pragma unroll
      for (int o = 0; o < kOut; ++o) out[(size_t)(b0 + m) * kOut + o] = 0.f;
    }
  }
}

extern "C" void kernel_launch(void* const* d_in, const int* in_sizes, int n_in,
                              void* d_out, int out_size, void* d_ws, size_t ws_size,
                              hipStream_t stream) {
  (void)n_in; (void)out_size; (void)d_ws; (void)ws_size;
  const float* x  = (const float*)d_in[0];
  const float* Wh = (const float*)d_in[1];
  const float* bh = (const float*)d_in[2];
  const float* Wo = (const float*)d_in[3];
  const float* bo = (const float*)d_in[4];
  float* out = (float*)d_out;

  const int B  = in_sizes[0] / (kT * kIn);   // 4096
  const int nb = B / 16;                     // 2 waves (64 threads) per 16-row tile
  hipLaunchKernelGGL(rnn_wmma_kernel, dim3(nb), dim3(64), 0, stream,
                     x, Wh, bh, Wo, bo, out);
}
